// network_30425548324978
// MI455X (gfx1250) — compile-verified, matrix operand order checked by reading, not run on hardware
//
#include <hip/hip_runtime.h>
#include <hip/hip_bf16.h>
#include <math.h>

typedef _Float16 f16;
typedef __attribute__((ext_vector_type(16))) _Float16 v16h;
typedef __attribute__((ext_vector_type(8)))  _Float16 v8h;
typedef __attribute__((ext_vector_type(8)))  float    v8f;

#define NNODES 50000
#define NPAD   50016      // 50000 + 16, a multiple of 16
#define NEDGE  800000
#define FDIM   128
#define NHEAD  4
#define CDIM   32
#define HC     128
#define NGRAPH 64
#define EDIM   16
#define EDPAD  32
#define LINSZ  256

// ---------------- ordered-float encoding for atomic max ----------------
__device__ __forceinline__ unsigned f2ord(float f) {
    unsigned u = __float_as_uint(f);
    return (u >> 31) ? ~u : (u | 0x80000000u);
}
__device__ __forceinline__ float ord2f(unsigned u) {
    u = (u >> 31) ? (u & 0x7fffffffu) : ~u;
    return __uint_as_float(u);
}

// ---------------- fills ----------------
__global__ void fill_f32_kernel(float* p, float v, long long n) {
    long long i = (long long)blockIdx.x * blockDim.x + threadIdx.x;
    if (i < n) p[i] = v;
}
__global__ void fill_u32_kernel(unsigned* p, unsigned v, long long n) {
    long long i = (long long)blockIdx.x * blockDim.x + threadIdx.x;
    if (i < n) p[i] = v;
}

// ---------------- f32 -> f16 convert with row/col zero padding ----------------
__global__ void cvt_pad_kernel(const float* __restrict__ src, f16* __restrict__ dst,
                               int rows, int kin, int rowsPad, int kpad) {
    long long i = (long long)blockIdx.x * blockDim.x + threadIdx.x;
    long long total = (long long)rowsPad * kpad;
    if (i >= total) return;
    int r = (int)(i / kpad), c = (int)(i % kpad);
    float v = (r < rows && c < kin) ? src[(long long)r * kin + c] : 0.0f;
    dst[i] = (f16)v;
}

// ---------------- embedding lookup ----------------
__global__ void embed_kernel(const int* __restrict__ x, const float* __restrict__ emb,
                             float* __restrict__ h, int n) {
    long long i = (long long)blockIdx.x * blockDim.x + threadIdx.x;
    if (i >= (long long)n * FDIM) return;
    int r = (int)(i >> 7), c = (int)(i & 127);
    h[i] = emb[x[r] * FDIM + c];
}

// ---------------- generic WMMA GEMM: out = A[M,K] * B[Nout,K]^T (+bias)(relu)(f16 store)
// Each wave computes a 16 x (NT*16) strip: A tile loaded once per K-step, NT WMMAs share it.
template <int NT>
__global__ __launch_bounds__(256)
void gemm_wmma_kernel(const f16* __restrict__ A, const f16* __restrict__ B,
                      const float* __restrict__ bias, void* __restrict__ Out,
                      int M, int K, int Nout, int flags) {
    int wave    = blockIdx.x * (blockDim.x >> 5) + (threadIdx.x >> 5);
    int ngroups = Nout >> (4 + (NT == 4 ? 2 : 1));     // Nout / (NT*16)
    int mtiles  = (M + 15) >> 4;
    if (wave >= mtiles * ngroups) return;
    int mt = wave / ngroups, ng = wave % ngroups;
    int lane = threadIdx.x & 31;
    int hlf = lane >> 4, r = lane & 15;
    const f16* arow = A + (long long)(mt * 16 + r) * K;            // A padded to mtiles*16 rows
    const f16* bbase = B + (long long)(ng * NT * 16 + r) * K;
    v8f c[NT];
#pragma unroll
    for (int t = 0; t < NT; t++) c[t] = (v8f){};
    for (int kk = 0; kk < K; kk += 32) {
        v8h a0 = *(const v8h*)(arow + kk + hlf * 8);
        v8h a1 = *(const v8h*)(arow + kk + 16 + hlf * 8);
        v16h a;
#pragma unroll
        for (int i = 0; i < 8; i++) { a[i] = a0[i]; a[i + 8] = a1[i]; }
#pragma unroll
        for (int t = 0; t < NT; t++) {
            const f16* brow = bbase + (long long)(t * 16) * K;
            v8h b0 = *(const v8h*)(brow + kk + hlf * 8);
            v8h b1 = *(const v8h*)(brow + kk + 16 + hlf * 8);
            v16h b;
#pragma unroll
            for (int i = 0; i < 8; i++) { b[i] = b0[i]; b[i + 8] = b1[i]; }
            c[t] = __builtin_amdgcn_wmma_f32_16x16x32_f16(false, a, false, b, (short)0, c[t], false, false);
        }
    }
#pragma unroll
    for (int t = 0; t < NT; t++) {
        int n = (ng * NT + t) * 16 + r;
        float bv = bias ? bias[n] : 0.0f;
#pragma unroll
        for (int i = 0; i < 8; i++) {
            int m = mt * 16 + i + 8 * hlf;
            if (m < M) {
                float o = c[t][i] + bv;
                if (flags & 1) o = fmaxf(o, 0.0f);
                if (flags & 2) ((f16*)Out)[(long long)m * Nout + n] = (f16)o;
                else           ((float*)Out)[(long long)m * Nout + n] = o;
            }
        }
    }
}

// ---------------- edge attention passes ----------------
__global__ void alpha_kernel(const float* __restrict__ q, const float* __restrict__ k,
                             const f16* __restrict__ ep, const int* __restrict__ src,
                             const int* __restrict__ dst, float* __restrict__ alpha,
                             unsigned* __restrict__ amax) {
    long long i = (long long)blockIdx.x * blockDim.x + threadIdx.x;
    if (i >= (long long)NEDGE * NHEAD) return;
    int e = (int)(i >> 2), hh = (int)(i & 3);
    int s = src[e], d = dst[e];
    const float* qp = q + (long long)d * HC + hh * CDIM;
    const float* kp = k + (long long)s * HC + hh * CDIM;
    const f16*  epp = ep + (long long)e * HC + hh * CDIM;
    float acc = 0.f;
    for (int c = 0; c < CDIM; c++) acc += qp[c] * (kp[c] + (float)epp[c]);
    acc *= 0.17677669529663689f;   // 1/sqrt(32)
    alpha[i] = acc;
    atomicMax(&amax[(long long)d * NHEAD + hh], f2ord(acc));
}

__global__ void exp_kernel(float* __restrict__ alpha, const unsigned* __restrict__ amax,
                           const int* __restrict__ dst, float* __restrict__ den) {
    long long i = (long long)blockIdx.x * blockDim.x + threadIdx.x;
    if (i >= (long long)NEDGE * NHEAD) return;
    int e = (int)(i >> 2), hh = (int)(i & 3);
    int d = dst[e];
    float ex = __expf(alpha[i] - ord2f(amax[(long long)d * NHEAD + hh]));
    alpha[i] = ex;
    atomicAdd(&den[(long long)d * NHEAD + hh], ex);
}

__global__ void message_kernel(const float* __restrict__ v, const f16* __restrict__ ep,
                               const float* __restrict__ ex, const float* __restrict__ den,
                               const int* __restrict__ src, const int* __restrict__ dst,
                               float* __restrict__ agg) {
    long long i = (long long)blockIdx.x * blockDim.x + threadIdx.x;
    if (i >= (long long)NEDGE * HC) return;
    int e = (int)(i >> 7), c = (int)(i & 127), hh = c >> 5;
    int s = src[e], d = dst[e];
    float a = ex[(long long)e * NHEAD + hh] / (den[(long long)d * NHEAD + hh] + 1e-16f);
    atomicAdd(&agg[(long long)d * HC + c], (v[(long long)s * HC + c] + (float)ep[i]) * a);
}

// ---------------- beta gating ----------------
__global__ void combine_kernel(const float* __restrict__ agg, const float* __restrict__ skip,
                               const float* __restrict__ Wb, float* __restrict__ out, int n) {
    int nd = blockIdx.x * blockDim.x + threadIdx.x;
    if (nd >= n) return;
    const float* o = agg + (long long)nd * HC;
    const float* s = skip + (long long)nd * HC;
    float bl = 0.f;
    for (int c = 0; c < HC; c++) {
        float ov = o[c], sv = s[c];
        bl += Wb[c] * ov + Wb[HC + c] * sv + Wb[2 * HC + c] * (ov - sv);
    }
    float beta = 1.f / (1.f + __expf(-bl));
    float* dst = out + (long long)nd * HC;
    for (int c = 0; c < HC; c++) dst[c] = beta * s[c] + (1.f - beta) * o[c];
}

// ---------------- batch norm ----------------
__global__ void bn_stats_kernel(const float* __restrict__ h, float* __restrict__ stats, int n) {
    __shared__ float ssum[CDIM], ssq[CDIM];
    if (threadIdx.x < CDIM) { ssum[threadIdx.x] = 0.f; ssq[threadIdx.x] = 0.f; }
    __syncthreads();
    long long i = (long long)blockIdx.x * blockDim.x + threadIdx.x;
    if (i < (long long)n * CDIM) {
        int c = (int)(i & (CDIM - 1));
        float v = h[i];
        atomicAdd(&ssum[c], v);
        atomicAdd(&ssq[c], v * v);
    }
    __syncthreads();
    if (threadIdx.x < CDIM) {
        atomicAdd(&stats[threadIdx.x], ssum[threadIdx.x]);
        atomicAdd(&stats[CDIM + threadIdx.x], ssq[threadIdx.x]);
    }
}
__global__ void bn_apply_kernel(const float* __restrict__ hin, const float* __restrict__ stats,
                                const float* __restrict__ g, const float* __restrict__ b,
                                float* __restrict__ hout, int n) {
    long long i = (long long)blockIdx.x * blockDim.x + threadIdx.x;
    if (i >= (long long)n * CDIM) return;
    int c = (int)(i & (CDIM - 1));
    float mean = stats[c] / (float)n;
    float var  = stats[CDIM + c] / (float)n - mean * mean;
    hout[i] = (hin[i] - mean) * rsqrtf(var + 1e-5f) * g[c] + b[c];
}

// ---------------- pooling ----------------
__global__ void count_kernel(const int* __restrict__ batch, float* __restrict__ counts, int n) {
    int i = blockIdx.x * blockDim.x + threadIdx.x;
    if (i < n) atomicAdd(&counts[batch[i]], 1.0f);
}
__global__ void pool_kernel(const float* __restrict__ h, const int* __restrict__ batch,
                            unsigned* __restrict__ gmax, float* __restrict__ gsum, int n) {
    long long i = (long long)blockIdx.x * blockDim.x + threadIdx.x;
    if (i >= (long long)n * CDIM) return;
    int nd = (int)(i >> 5), c = (int)(i & 31);
    int b = batch[nd];
    float v = h[i];
    atomicMax(&gmax[b * CDIM + c], f2ord(v));
    atomicAdd(&gsum[b * CDIM + c], v);
}
__global__ void rep_accum_kernel(const unsigned* __restrict__ gmax, const float* __restrict__ gsum,
                                 const float* __restrict__ counts, float* __restrict__ rep) {
    int i = blockIdx.x * blockDim.x + threadIdx.x;
    if (i >= NGRAPH * CDIM) return;
    int g = i >> 5, c = i & 31;
    float cnt = fmaxf(counts[g], 1.0f);
    rep[g * 2 * CDIM + c]        += ord2f(gmax[i]);
    rep[g * 2 * CDIM + CDIM + c] += gsum[i] / cnt;
}

// ---------------- final head ----------------
__global__ void head_kernel(const float* __restrict__ z2, const float* __restrict__ W3,
                            const float* __restrict__ b3, float* __restrict__ out) {
    int g = threadIdx.x;
    if (g >= NGRAPH) return;
    float acc = b3[0];
    for (int c = 0; c < LINSZ / 2; c++) acc += z2[g * (LINSZ / 2) + c] * W3[c];
    out[g] = 1.f / (1.f + __expf(-acc));
}

// =====================================================================
extern "C" void kernel_launch(void* const* d_in, const int* in_sizes, int n_in,
                              void* d_out, int out_size, void* d_ws, size_t ws_size,
                              hipStream_t stream) {
    const int*   x          = (const int*)d_in[0];
    const float* edge_attr  = (const float*)d_in[1];
    const int*   src        = (const int*)d_in[2];
    const int*   dstp       = ((const int*)d_in[2]) + NEDGE;
    const int*   batch      = (const int*)d_in[3];
    // bns: [4..9] (g,b) x3 ; convs: base 10,20,30 (Wb,We,Wk,Wq,Ws,Wv,bk,bq,bs,bv)
    const float* emb   = (const float*)d_in[40];
    const float* l1W   = (const float*)d_in[41];
    const float* l1b   = (const float*)d_in[42];
    const float* l2W   = (const float*)d_in[43];
    const float* l2b   = (const float*)d_in[44];
    const float* l3W   = (const float*)d_in[45];
    const float* l3b   = (const float*)d_in[46];

    // -------- workspace bump allocator --------
    char* base = (char*)d_ws;
    size_t off = 0;
    auto alloc = [&](size_t bytes) -> void* {
        void* r = base + off;
        off = (off + bytes + 255) & ~(size_t)255;
        return r;
    };
    float* hcur  = (float*)alloc((size_t)NNODES * HC * 4);
    f16*   h16   = (f16*)  alloc((size_t)NPAD * HC * 2);
    float* qb    = (float*)alloc((size_t)NNODES * HC * 4);
    float* kb    = (float*)alloc((size_t)NNODES * HC * 4);
    float* vb    = (float*)alloc((size_t)NNODES * HC * 4);
    float* sb    = (float*)alloc((size_t)NNODES * HC * 4);
    float* agg   = (float*)alloc((size_t)NNODES * HC * 4);
    float* hc    = (float*)alloc((size_t)NNODES * HC * 4);
    float* hl    = (float*)alloc((size_t)NNODES * CDIM * 4);
    f16*   ea16  = (f16*)  alloc((size_t)NEDGE * EDPAD * 2);
    f16*   ep16  = (f16*)  alloc((size_t)NEDGE * HC * 2);
    float* alpha = (float*)alloc((size_t)NEDGE * NHEAD * 4);
    unsigned* amax = (unsigned*)alloc((size_t)NNODES * NHEAD * 4);
    float* den   = (float*)alloc((size_t)NNODES * NHEAD * 4);
    float* stats = (float*)alloc(2 * CDIM * 4);
    float* counts= (float*)alloc(NGRAPH * 4);
    unsigned* gmax = (unsigned*)alloc(NGRAPH * CDIM * 4);
    float* gsum  = (float*)alloc(NGRAPH * CDIM * 4);
    float* rep   = (float*)alloc(NGRAPH * 2 * CDIM * 4);
    f16*   rep16 = (f16*)  alloc(NGRAPH * 2 * CDIM * 2);
    float* z1    = (float*)alloc(NGRAPH * LINSZ * 4);
    f16*   z1h   = (f16*)  alloc(NGRAPH * LINSZ * 2);
    float* z2    = (float*)alloc(NGRAPH * (LINSZ / 2) * 4);
    f16*   wq16  = (f16*)alloc((size_t)HC * FDIM * 2);
    f16*   wk16  = (f16*)alloc((size_t)HC * FDIM * 2);
    f16*   wv16  = (f16*)alloc((size_t)HC * FDIM * 2);
    f16*   ws16  = (f16*)alloc((size_t)HC * FDIM * 2);
    f16*   we16  = (f16*)alloc((size_t)HC * EDPAD * 2);
    f16*   lw16  = (f16*)alloc((size_t)CDIM * HC * 2);
    f16*   l1w16 = (f16*)alloc((size_t)LINSZ * 2 * CDIM * 2);
    f16*   l2w16 = (f16*)alloc((size_t)(LINSZ / 2) * LINSZ * 2);
    (void)ws_size; (void)n_in; (void)in_sizes; (void)out_size;

    auto blocks = [](long long n, int bs) { return (unsigned)((n + bs - 1) / bs); };
    auto fillf = [&](float* p, float v, long long n) {
        fill_f32_kernel<<<blocks(n, 256), 256, 0, stream>>>(p, v, n);
    };
    auto fillu = [&](unsigned* p, unsigned v, long long n) {
        fill_u32_kernel<<<blocks(n, 256), 256, 0, stream>>>(p, v, n);
    };
    auto cvt = [&](const float* s, f16* d, int rows, int kin, int rowsPad, int kpad) {
        long long n = (long long)rowsPad * kpad;
        cvt_pad_kernel<<<blocks(n, 256), 256, 0, stream>>>(s, d, rows, kin, rowsPad, kpad);
    };
    auto gemm = [&](const f16* A, const f16* B, const float* bias, void* out,
                    int M, int K, int Nout, int flags) {
        int mtiles = (M + 15) / 16;
        if (Nout % 64 == 0) {
            long long tiles = (long long)mtiles * (Nout / 64);
            gemm_wmma_kernel<4><<<blocks(tiles, 8), 256, 0, stream>>>(A, B, bias, out, M, K, Nout, flags);
        } else {
            long long tiles = (long long)mtiles * (Nout / 32);
            gemm_wmma_kernel<2><<<blocks(tiles, 8), 256, 0, stream>>>(A, B, bias, out, M, K, Nout, flags);
        }
    };

    const unsigned NEGMAX = 0x00800000u;   // f2ord(-FLT_MAX)

    // ---- setup ----
    embed_kernel<<<blocks((long long)NNODES * FDIM, 256), 256, 0, stream>>>(x, emb, hcur, NNODES);
    cvt(edge_attr, ea16, NEDGE, EDIM, NEDGE, EDPAD);
    fillf(counts, 0.f, NGRAPH);
    count_kernel<<<blocks(NNODES, 256), 256, 0, stream>>>(batch, counts, NNODES);
    fillf(rep, 0.f, NGRAPH * 2 * CDIM);

    // ---- conv stack ----
    for (int l = 0; l < 3; l++) {
        int cb  = 10 + 10 * l;                         // conv param base index
        int din = (l == 0) ? FDIM : CDIM;
        const float* Wb = (const float*)d_in[cb + 0];
        const float* We = (const float*)d_in[cb + 1];
        const float* Wk = (const float*)d_in[cb + 2];
        const float* Wq = (const float*)d_in[cb + 3];
        const float* Ws = (const float*)d_in[cb + 4];
        const float* Wv = (const float*)d_in[cb + 5];
        const float* bk = (const float*)d_in[cb + 6];
        const float* bq = (const float*)d_in[cb + 7];
        const float* bs = (const float*)d_in[cb + 8];
        const float* bv = (const float*)d_in[cb + 9];
        const float* linW = (const float*)d_in[47 + 2 * l];
        const float* linb = (const float*)d_in[48 + 2 * l];
        const float* bng  = (const float*)d_in[4 + 2 * l];
        const float* bnb  = (const float*)d_in[5 + 2 * l];

        // projections (WMMA)
        cvt(hcur, h16, NNODES, din, NPAD, din);
        cvt(Wq, wq16, HC, din, HC, din);
        cvt(Wk, wk16, HC, din, HC, din);
        cvt(Wv, wv16, HC, din, HC, din);
        cvt(Ws, ws16, HC, din, HC, din);
        cvt(We, we16, HC, EDIM, HC, EDPAD);
        gemm(h16, wq16, bq, qb, NNODES, din, HC, 0);
        gemm(h16, wk16, bk, kb, NNODES, din, HC, 0);
        gemm(h16, wv16, bv, vb, NNODES, din, HC, 0);
        gemm(h16, ws16, bs, sb, NNODES, din, HC, 0);
        gemm(ea16, we16, nullptr, ep16, NEDGE, EDPAD, HC, 2);   // f16 store

        // edge softmax + aggregation
        fillu(amax, NEGMAX, (long long)NNODES * NHEAD);
        fillf(den, 0.f, (long long)NNODES * NHEAD);
        fillf(agg, 0.f, (long long)NNODES * HC);
        alpha_kernel<<<blocks((long long)NEDGE * NHEAD, 256), 256, 0, stream>>>(qb, kb, ep16, src, dstp, alpha, amax);
        exp_kernel<<<blocks((long long)NEDGE * NHEAD, 256), 256, 0, stream>>>(alpha, amax, dstp, den);
        message_kernel<<<blocks((long long)NEDGE * HC, 256), 256, 0, stream>>>(vb, ep16, alpha, den, src, dstp, agg);
        combine_kernel<<<blocks(NNODES, 256), 256, 0, stream>>>(agg, sb, Wb, hc, NNODES);

        // lin + relu (WMMA) + batchnorm
        cvt(hc, h16, NNODES, HC, NPAD, HC);
        cvt(linW, lw16, CDIM, HC, CDIM, HC);
        gemm(h16, lw16, linb, hl, NNODES, HC, CDIM, 1);
        fillf(stats, 0.f, 2 * CDIM);
        bn_stats_kernel<<<blocks((long long)NNODES * CDIM, 256), 256, 0, stream>>>(hl, stats, NNODES);
        bn_apply_kernel<<<blocks((long long)NNODES * CDIM, 256), 256, 0, stream>>>(hl, stats, bng, bnb, hcur, NNODES);

        // pooling (every layer after the first)
        if (l >= 1) {
            fillu(gmax, NEGMAX, NGRAPH * CDIM);
            fillf(gsum, 0.f, NGRAPH * CDIM);
            pool_kernel<<<blocks((long long)NNODES * CDIM, 256), 256, 0, stream>>>(hcur, batch, gmax, gsum, NNODES);
            rep_accum_kernel<<<blocks(NGRAPH * CDIM, 256), 256, 0, stream>>>(gmax, gsum, counts, rep);
        }
    }

    // ---- MLP head ----
    cvt(rep, rep16, NGRAPH, 2 * CDIM, NGRAPH, 2 * CDIM);
    cvt(l1W, l1w16, LINSZ, 2 * CDIM, LINSZ, 2 * CDIM);
    gemm(rep16, l1w16, l1b, z1, NGRAPH, 2 * CDIM, LINSZ, 1);
    cvt(z1, z1h, NGRAPH, LINSZ, NGRAPH, LINSZ);
    cvt(l2W, l2w16, LINSZ / 2, LINSZ, LINSZ / 2, LINSZ);
    gemm(z1h, l2w16, l2b, z2, NGRAPH, LINSZ, LINSZ / 2, 1);
    head_kernel<<<1, 64, 0, stream>>>(z2, l3W, l3b, (float*)d_out);
}
